// MultiRelGATLayer_78237124264218
// MI455X (gfx1250) — matrix operand order repaired and checked by reading, not compiled, not run on hardware
//
#include <hip/hip_runtime.h>

typedef __attribute__((ext_vector_type(16))) __bf16 v16bf;
typedef __attribute__((ext_vector_type(8)))  float  v8f;
typedef __attribute__((ext_vector_type(4)))  float  v4f;

#define B_   2
#define N_   2048
#define D_   128
#define R_   3
#define HH_  4
#define HD_  32
#define RBH_ (R_*B_*HH_)      // 24
#define NW_  (N_/32)          // 64 mask words per row
#define NT_  (N_/16)          // 128 i-tiles per (r,b,h)

__device__ __forceinline__ float lrelu(float x) { return x > 0.f ? x : 0.2f * x; }

// ---------------------------------------------------------------------------
// Phase 1: Wh[rbh][n][f] = sum_d H[b][n][d] * W[r][h][d][f]   (bf16 WMMA)
// one wave per 16-row tile of one (r,b,h)
// ---------------------------------------------------------------------------
__global__ void k_wh(const float* __restrict__ H, const float* __restrict__ W,
                     float* __restrict__ wh) {
  int wid  = (blockIdx.x * blockDim.x + threadIdx.x) >> 5;   // 0..3071
  int lane = threadIdx.x & 31;
  int it   = wid & (NT_ - 1);
  int rbh  = wid >> 7;
  int r    = rbh / (B_ * HH_);
  int b    = (rbh / HH_) % B_;
  int h    = rbh % HH_;
  int m    = lane & 15, half = lane >> 4;

  const float* hrow = H + ((size_t)b * N_ + it * 16 + m) * D_;
  const float* wmat = W + ((size_t)(r * HH_ + h)) * D_ * HD_;

  v8f c0 = {}; v8f c1 = {};
  for (int k0 = 0; k0 < D_; k0 += 32) {
    // A fragment: lane = row M=m; elements e<8 -> K=k0+8*half+e, e>=8 -> K=k0+16+8*half+(e-8)
    v16bf a;
    const float* runA = hrow + k0 + 8 * half;
    const float* runB = hrow + k0 + 16 + 8 * half;
#pragma unroll
    for (int e = 0; e < 8; ++e) { a[e] = (__bf16)runA[e]; a[e + 8] = (__bf16)runB[e]; }
    // B fragments: lane = col N=m (+16 for tile 1); e<8 -> K=k0+8*half+e, e>=8 -> K=k0+16+8*half+(e-8)
    v16bf b0, b1;
#pragma unroll
    for (int e = 0; e < 16; ++e) {
      int kk = k0 + ((e < 8) ? (8 * half + e) : (16 + 8 * half + (e - 8)));
      b0[e] = (__bf16)wmat[kk * HD_ + m];
      b1[e] = (__bf16)wmat[kk * HD_ + 16 + m];
    }
    c0 = __builtin_amdgcn_wmma_f32_16x16x32_bf16(false, a, false, b0, (short)0, c0, false, false);
    c1 = __builtin_amdgcn_wmma_f32_16x16x32_bf16(false, a, false, b1, (short)0, c1, false, false);
  }
  // D layout: lane col = m, rows = v + 8*half
  float* out = wh + ((size_t)rbh * N_ + it * 16) * HD_;
#pragma unroll
  for (int v = 0; v < 8; ++v) {
    int rr = v + 8 * half;
    out[rr * HD_ + m]      = c0[v];
    out[rr * HD_ + 16 + m] = c1[v];
  }
}

// ---------------------------------------------------------------------------
// Phase 1b: e_src / e_dst  (one thread per (rbh, n))
// ---------------------------------------------------------------------------
__global__ void k_e(const float* __restrict__ wh, const float* __restrict__ asrc,
                    const float* __restrict__ adst, float* __restrict__ esrc,
                    float* __restrict__ edst) {
  int idx = blockIdx.x * blockDim.x + threadIdx.x;
  if (idx >= RBH_ * N_) return;
  int rbh = idx / N_;
  int r = rbh / (B_ * HH_);
  int h = rbh % HH_;
  const float* row = wh + (size_t)idx * HD_;
  const float* as  = asrc + (size_t)(r * HH_ + h) * HD_;
  const float* ad  = adst + (size_t)(r * HH_ + h) * HD_;
  float s = 0.f, d = 0.f;
#pragma unroll
  for (int f = 0; f < HD_; ++f) { float w = row[f]; s += w * as[f]; d += w * ad[f]; }
  esrc[idx] = s; edst[idx] = d;
}

// ---------------------------------------------------------------------------
// Phase 1c: mdst[rbh] = max_n e_dst[rbh][n]
// ---------------------------------------------------------------------------
__global__ void k_mdst(const float* __restrict__ edst, float* __restrict__ mdst) {
  __shared__ float red[256];
  int rbh = blockIdx.x;
  float mx = -1e30f;
  for (int n = threadIdx.x; n < N_; n += 256) mx = fmaxf(mx, edst[rbh * N_ + n]);
  red[threadIdx.x] = mx; __syncthreads();
  for (int s = 128; s > 0; s >>= 1) {
    if (threadIdx.x < s) red[threadIdx.x] = fmaxf(red[threadIdx.x], red[threadIdx.x + s]);
    __syncthreads();
  }
  if (threadIdx.x == 0) mdst[rbh] = red[0];
}

// ---------------------------------------------------------------------------
// Phase 1d: repack Wh f32 -> bf16 in WMMA B-operand layout
// whb[rbh][jt][nt][lane][8 dwords];  element e of lane -> Wh[jt*32 + (e<8 ? 8*half+e : 16+8*half+(e-8))][nt*16 + m]
// ---------------------------------------------------------------------------
__global__ void k_repack(const float* __restrict__ wh, unsigned int* __restrict__ whb) {
  int wid  = (blockIdx.x * blockDim.x + threadIdx.x) >> 5;   // 24*64*2 = 3072
  int lane = threadIdx.x & 31;
  int nt   = wid & 1;
  int jt   = (wid >> 1) & 63;
  int rbh  = wid >> 7;
  int m    = lane & 15, half = lane >> 4;
  int f    = nt * 16 + m;
  v16bf bb;
#pragma unroll
  for (int e = 0; e < 16; ++e) {
    int j = jt * 32 + ((e < 8) ? (8 * half + e) : (16 + 8 * half + (e - 8)));
    bb[e] = (__bf16)wh[((size_t)rbh * N_ + j) * HD_ + f];
  }
  *(v16bf*)(whb + (((size_t)rbh * 64 + jt) * 2 + nt) * 256 + lane * 8) = bb;
}

// ---------------------------------------------------------------------------
// Phase 1e: pack adjacency to bitmask via wave32 ballot (100MB -> 3.1MB)
// ---------------------------------------------------------------------------
__global__ void k_packA(const int* __restrict__ A, unsigned int* __restrict__ ap) {
  size_t tid = (size_t)blockIdx.x * blockDim.x + threadIdx.x;
  unsigned long long bal = __ballot(A[tid] > 0);
  if ((threadIdx.x & 31) == 0) ap[tid >> 5] = (unsigned int)bal;
}

// ---------------------------------------------------------------------------
// Phase 2: fused masked softmax + aggregation.  One wave per (b,h,i-tile),
// relations accumulated in-wave (deterministic).  2 WMMAs per 32-wide j step.
// A-fragment K slots per lane are two contiguous 8-runs -> vector loads of
// e_dst, no cross-lane shuffles in the hot loop.
// ---------------------------------------------------------------------------
__global__ void k_attn(const unsigned int* __restrict__ ap,
                       const unsigned int* __restrict__ whb,
                       const float* __restrict__ esrc,
                       const float* __restrict__ edst,
                       const float* __restrict__ mdst,
                       float* __restrict__ agg) {
  int wid  = (blockIdx.x * blockDim.x + threadIdx.x) >> 5;   // B*HH*NT = 1024
  int lane = threadIdx.x & 31;
  int it   = wid & (NT_ - 1);
  int h    = (wid >> 7) & (HH_ - 1);
  int b    = wid >> 9;
  int m    = lane & 15, half = lane >> 4;
  int row  = it * 16 + m;

  v8f acc0 = {}; v8f acc1 = {};
  for (int r = 0; r < R_; ++r) {
    int rbh = (r * B_ + b) * HH_ + h;
    float es   = esrc[(size_t)rbh * N_ + row];
    float mrow = lrelu(es + mdst[rbh]);                // >= masked row max
    const unsigned int* aprow = ap + ((size_t)(r * B_ + b) * N_ + row) * NW_;
    const float* ed = edst + (size_t)rbh * N_ + 8 * half;
    const unsigned int* wb = whb + (size_t)rbh * 64 * 512;

    v8f c0 = {}; v8f c1 = {};
    float l = 0.f;
    for (int jt = 0; jt < NW_; ++jt) {
      unsigned int word = aprow[jt];
      const float* edt = ed + jt * 32;
      v4f e0 = *(const v4f*)(edt);                     // K = 8*half + 0..3
      v4f e1 = *(const v4f*)(edt + 4);                 // K = 8*half + 4..7
      v4f e2 = *(const v4f*)(edt + 16);                // K = 16 + 8*half + 0..3
      v4f e3 = *(const v4f*)(edt + 20);                // K = 16 + 8*half + 4..7
      unsigned int w0 = word >> (8 * half);
      unsigned int w1 = word >> (16 + 8 * half);
      float ev[16];
#pragma unroll
      for (int k = 0; k < 4; ++k) {
        ev[k] = e0[k]; ev[4 + k] = e1[k]; ev[8 + k] = e2[k]; ev[12 + k] = e3[k];
      }
      v16bf pa;
#pragma unroll
      for (int e = 0; e < 16; ++e) {
        unsigned int bit = (e < 8) ? ((w0 >> e) & 1u) : ((w1 >> (e - 8)) & 1u);
        float pe = bit ? __expf(lrelu(es + ev[e]) - mrow) : 0.f;
        pa[e] = (__bf16)pe;
        l += pe;
      }
      const unsigned int* base = wb + (size_t)jt * 512 + lane * 8;
      v16bf b0 = *(const v16bf*)(base);
      v16bf b1 = *(const v16bf*)(base + 256);
      c0 = __builtin_amdgcn_wmma_f32_16x16x32_bf16(false, pa, false, b0, (short)0, c0, false, false);
      c1 = __builtin_amdgcn_wmma_f32_16x16x32_bf16(false, pa, false, b1, (short)0, c1, false, false);
    }
    l += __shfl_xor(l, 16, 32);                        // combine the two K-halves of row m
#pragma unroll
    for (int v = 0; v < 8; ++v) {
      float lr  = __shfl(l, v + 8 * half, 32);         // row sum for accumulator row
      float inv = (lr > 0.f) ? 1.f / lr : 0.f;         // empty neighborhood -> 0
      acc0[v] += c0[v] * inv;
      acc1[v] += c1[v] * inv;
    }
  }
  const float invR = 1.f / (float)R_;
  float* out = agg + ((size_t)b * N_ + it * 16) * D_ + h * HD_;
#pragma unroll
  for (int v = 0; v < 8; ++v) {
    int rr = v + 8 * half;
    out[rr * D_ + m]      = acc0[v] * invR;
    out[rr * D_ + 16 + m] = acc1[v] * invR;
  }
}

// ---------------------------------------------------------------------------
// Phase 3: residual + LayerNorm, one wave per row (4 elems/lane, shuffle reduce)
// ---------------------------------------------------------------------------
__global__ void k_ln(const float* __restrict__ H, const float* __restrict__ agg,
                     const float* __restrict__ gamma, const float* __restrict__ beta,
                     float* __restrict__ out) {
  int wid  = (blockIdx.x * blockDim.x + threadIdx.x) >> 5;   // B*N = 4096
  int lane = threadIdx.x & 31;
  if (wid >= B_ * N_) return;
  v4f hv = *(const v4f*)(H   + (size_t)wid * D_ + lane * 4);
  v4f av = *(const v4f*)(agg + (size_t)wid * D_ + lane * 4);
  float x[4]; float s = 0.f, s2 = 0.f;
#pragma unroll
  for (int k = 0; k < 4; ++k) { x[k] = hv[k] + av[k]; s += x[k]; s2 += x[k] * x[k]; }
#pragma unroll
  for (int off = 16; off > 0; off >>= 1) {
    s  += __shfl_xor(s,  off, 32);
    s2 += __shfl_xor(s2, off, 32);
  }
  float mu   = s * (1.f / D_);
  float var  = s2 * (1.f / D_) - mu * mu;
  float rstd = rsqrtf(var + 1e-5f);
#pragma unroll
  for (int k = 0; k < 4; ++k) {
    int f = lane * 4 + k;
    out[(size_t)wid * D_ + f] = (x[k] - mu) * rstd * gamma[f] + beta[f];
  }
}

extern "C" void kernel_launch(void* const* d_in, const int* in_sizes, int n_in,
                              void* d_out, int out_size, void* d_ws, size_t ws_size,
                              hipStream_t stream) {
  const float* H     = (const float*)d_in[0];
  const int*   A     = (const int*)  d_in[1];
  const float* W     = (const float*)d_in[2];
  const float* a_src = (const float*)d_in[3];
  const float* a_dst = (const float*)d_in[4];
  const float* gamma = (const float*)d_in[5];
  const float* beta  = (const float*)d_in[6];
  (void)in_sizes; (void)n_in; (void)out_size; (void)ws_size;

  float* ws = (float*)d_ws;
  // layout (4-byte elements, 256B-aligned offsets):
  float*        whf   = ws;                              // 1,572,864  (dead after repack)
  unsigned int* whb   = (unsigned int*)(ws + 1572864);   //   786,432
  float*        esrc  = ws + 2359296;                    //    49,152
  float*        edst  = ws + 2408448;                    //    49,152
  float*        md    = ws + 2457600;                    //        64
  unsigned int* apack = (unsigned int*)ws;               //   786,432 (overlays dead whf)
  float*        agg   = ws + 786432;                     //   524,288 (overlays dead whf)

  k_wh    <<<768,   128, 0, stream>>>(H, W, whf);
  k_e     <<<192,   256, 0, stream>>>(whf, a_src, a_dst, esrc, edst);
  k_mdst  <<<RBH_,  256, 0, stream>>>(edst, md);
  k_repack<<<768,   128, 0, stream>>>(whf, whb);
  k_packA <<<98304, 256, 0, stream>>>(A, apack);         // runs after whf is dead
  k_attn  <<<128,   256, 0, stream>>>(apack, whb, esrc, edst, md, agg);
  k_ln    <<<512,   256, 0, stream>>>(H, agg, gamma, beta, (float*)d_out);
}
